// SelfAttention_60644938219611
// MI455X (gfx1250) — matrix operand order repaired and checked by reading, not compiled, not run on hardware
//
#include <hip/hip_runtime.h>

// ---------------------------------------------------------------------------
// Self-attention (B=4, N=4096, D=256, f32, no 1/sqrt(d) scaling) for gfx1250.
// bf16 WMMA for all matmuls, flash-style online softmax (no 268MB score
// matrix), Tensor Data Mover (TENSOR_LOAD_TO_LDS) for K/V block staging with
// double-buffered K and DMA/compute overlap via s_wait_tensorcnt.
// ---------------------------------------------------------------------------

typedef __attribute__((ext_vector_type(16))) __bf16 v16bf;
typedef __attribute__((ext_vector_type(8)))  __bf16 v8bf;
typedef __attribute__((ext_vector_type(8)))  float  v8f;
typedef __attribute__((ext_vector_type(4)))  float  f32x4;
typedef __attribute__((ext_vector_type(4)))  unsigned int u32x4;
typedef __attribute__((ext_vector_type(8)))  int    i32x8;
typedef __attribute__((ext_vector_type(4)))  int    i32x4;

#define DIM      256
#define B_BATCH  4
#define NSEQ     4096
#define MROWS    (B_BATCH * NSEQ)   // 16384
#define GK       32                 // K-step for GEMM
#define GROWS    64                 // rows per GEMM workgroup
#define BR       64                 // queries per attention workgroup
#define BC       32                 // keys per block

__device__ __forceinline__ __bf16 f32_to_bf16(float f) {
  union { float f; unsigned u; } a; a.f = f;
  unsigned r = a.u + 0x7FFFu + ((a.u >> 16) & 1u);   // round-to-nearest-even
  union { unsigned short s; __bf16 b; } o;
  o.s = (unsigned short)(r >> 16);
  return o.b;
}

// A-matrix fragment (16x32 bf16), CDNA5 ISA 7.12.2:
// lanes 0-15 (row = lane): k 0..7 then 16..23; lanes 16-31: k 8..15 then 24..31.
__device__ __forceinline__ v16bf load_frag_a(const __bf16* __restrict__ base,
                                             int stride, int lane) {
  const int r = lane & 15, h = lane >> 4;
  const __bf16* p = base + (size_t)r * stride;
  union { v16bf v; struct { v8bf lo; v8bf hi; } s; } u;
  u.s.lo = *(const v8bf*)(p + (h ? 8 : 0));
  u.s.hi = *(const v8bf*)(p + (h ? 24 : 16));
  return u.v;
}

// B-matrix fragment (32x16 bf16): column n = lane&15; lane-half h holds
// k 8h..8h+7 then 16+8h..16+8h+7. base -> Bt[n0][0], Bt is [n][k].
__device__ __forceinline__ v16bf load_frag_b(const __bf16* __restrict__ base,
                                             size_t stride, int lane) {
  const int n = lane & 15, h = lane >> 4;
  const __bf16* p = base + (size_t)n * stride;
  union { v16bf v; struct { v8bf lo; v8bf hi; } s; } u;
  u.s.lo = *(const v8bf*)(p + (h ? 8 : 0));
  u.s.hi = *(const v8bf*)(p + (h ? 24 : 16));
  return u.v;
}

// ---------------------------------------------------------------------------
// Tensor Data Mover: 2D bf16 tile (tile_d0 contiguous elems per row,
// tile_d1 rows, row stride stride0 elems) global -> LDS (rows packed).
// Descriptor per CDNA5 ISA ch.8. Call from uniform control flow (one wave).
// ---------------------------------------------------------------------------
__device__ __forceinline__ void tdm_load_2d_bf16(
    unsigned lds_addr, const void* gsrc,
    unsigned tile_d0, unsigned tile_d1,
    unsigned tensor_d0, unsigned tensor_d1,
    unsigned long long stride0_elems)
{
  unsigned long long ga = (unsigned long long)(size_t)gsrc;
  u32x4 g0;
  g0[0] = 1u;                                   // count=1 (valid), user mode
  g0[1] = lds_addr;                             // LDS byte address
  g0[2] = (unsigned)ga;                         // global_addr[31:0]
  g0[3] = (unsigned)((ga >> 32) & 0x01FFFFFFu)  // global_addr[56:32]
        | (2u << 30);                           // type = 2 ("image")
  i32x8 g1;
  g1[0] = (int)(1u << 16);                      // wg_mask=0, data_size=1 (2B)
  g1[1] = (int)((tensor_d0 & 0xFFFFu) << 16);   // tensor_dim0[15:0]
  g1[2] = (int)(((tensor_d0 >> 16) & 0xFFFFu)   // tensor_dim0[31:16]
        |       ((tensor_d1 & 0xFFFFu) << 16)); // tensor_dim1[15:0]
  g1[3] = (int)(((tensor_d1 >> 16) & 0xFFFFu)   // tensor_dim1[31:16]
        |       ((tile_d0 & 0xFFFFu) << 16));   // tile_dim0
  g1[4] = (int)(tile_d1 & 0xFFFFu);             // tile_dim1 (tile_dim2=0)
  g1[5] = (int)(unsigned)(stride0_elems & 0xFFFFFFFFu);     // dim0_stride lo
  g1[6] = (int)(unsigned)((stride0_elems >> 32) & 0xFFFFu); // stride hi
  g1[7] = 0;
  i32x4 gz = {0, 0, 0, 0};
#if __clang_major__ >= 23
  i32x8 gz8 = {};
  __builtin_amdgcn_tensor_load_to_lds(g0, g1, gz, gz, gz8, 0);
#else
  __builtin_amdgcn_tensor_load_to_lds(g0, g1, gz, gz, 0);
#endif
}

// ---------------------------------------------------------------------------
// C[M,256] = X[M,256]*W[256,256] + b   (f32 in -> bf16 WMMA -> f32 accum)
// MODE: 0 = f32 row-major, 1 = bf16 row-major,
//       2 = bf16 transposed per batch: out[(b*DIM+n)*NSEQ + (row%NSEQ)]
// Compile-time MODE -> branch-free epilogue.
// ---------------------------------------------------------------------------
template <int MODE>
__global__ __launch_bounds__(256) void gemm_bias_kernel(
    const float* __restrict__ X, const float* __restrict__ W,
    const float* __restrict__ bias, void* __restrict__ out, int M)
{
  __shared__ __bf16 Xs[GROWS][GK];   // 4 KB  [row][k]
  __shared__ __bf16 Wt[DIM][GK];     // 16 KB [n][k] (transposed)

  const int tid  = threadIdx.x;
  const int wave = tid >> 5, lane = tid & 31;
  const int half = lane >> 4, nlane = lane & 15;
  const int rowBase = blockIdx.x * GROWS;
  if (rowBase >= M) return;
  const int wr = (wave & 3) * 16;
  const int wc = (wave >> 2) * 128;

  v8f acc[8] = {};

  for (int k0 = 0; k0 < DIM; k0 += GK) {
    __syncthreads();
    // X tile: 64x32 f32, float4 loads, packed b64 bf16 stores. (2 iters)
#pragma unroll
    for (int ii = 0; ii < 2; ++ii) {
      int i = tid + ii * 256;
      int r = i >> 3, c4 = (i & 7) * 4;
      f32x4 x = *(const f32x4*)&X[(size_t)(rowBase + r) * DIM + k0 + c4];
      union { __bf16 h[4]; unsigned long long u; } pk;
      pk.h[0] = f32_to_bf16(x[0]); pk.h[1] = f32_to_bf16(x[1]);
      pk.h[2] = f32_to_bf16(x[2]); pk.h[3] = f32_to_bf16(x[3]);
      *(unsigned long long*)&Xs[r][c4] = pk.u;
    }
    // W tile: rows k0..k0+31 x 256 cols, float4 loads, transposed stores. (8)
#pragma unroll
    for (int ii = 0; ii < 8; ++ii) {
      int i = tid + ii * 256;
      int r = i >> 6, c4 = (i & 63) * 4;
      f32x4 w = *(const f32x4*)&W[(size_t)(k0 + r) * DIM + c4];
      Wt[c4 + 0][r] = f32_to_bf16(w[0]);
      Wt[c4 + 1][r] = f32_to_bf16(w[1]);
      Wt[c4 + 2][r] = f32_to_bf16(w[2]);
      Wt[c4 + 3][r] = f32_to_bf16(w[3]);
    }
    __syncthreads();
    v16bf a = load_frag_a(&Xs[wr][0], GK, lane);
#pragma unroll
    for (int t = 0; t < 8; ++t) {
      v16bf bw = load_frag_b(&Wt[wc + 16 * t][0], GK, lane);
      acc[t] = __builtin_amdgcn_wmma_f32_16x16x32_bf16(
          false, a, false, bw, (short)0, acc[t], false, false);
    }
  }

#pragma unroll
  for (int t = 0; t < 8; ++t) {
    const int n = wc + 16 * t + nlane;
    const float bv = bias[n];
    if (MODE == 2) {
      // Transposed: this lane's 8 values are 8 consecutive q positions
      // on row (b*DIM + n)  ->  one aligned 16B store.
      const size_t r0 = (size_t)(rowBase + wr + 8 * half);
      const size_t bb = r0 / NSEQ, q0 = r0 % NSEQ;
      union { __bf16 h[8]; v8bf v; } pk;
#pragma unroll
      for (int v = 0; v < 8; ++v) pk.h[v] = f32_to_bf16(acc[t][v] + bv);
      *(v8bf*)&((__bf16*)out)[(bb * DIM + n) * NSEQ + q0] = pk.v;
    } else {
#pragma unroll
      for (int v = 0; v < 8; ++v) {
        size_t r = (size_t)(rowBase + wr + v + 8 * half);
        float val = acc[t][v] + bv;
        if (MODE == 0) ((float*)out)[r * DIM + n] = val;
        else           ((__bf16*)out)[r * DIM + n] = f32_to_bf16(val);
      }
    }
  }
}

// ---------------------------------------------------------------------------
// Flash attention. One workgroup = 64 queries of one batch, 8 waves, each
// owning a 16x128 slice of the 64x256 output accumulator.
// K blocks: TDM double-buffered.  V blocks: TDM from pre-transposed V^T,
// overlapped with the S-matmul + softmax phase.
// ---------------------------------------------------------------------------
__global__ __launch_bounds__(256) void attn_kernel(
    const __bf16* __restrict__ Qb, const __bf16* __restrict__ Kb,
    const __bf16* __restrict__ Vtg,   // V^T: [b*DIM + d][NSEQ]
    float* __restrict__ ctx)
{
  __shared__ __bf16 Ks[2][BC][DIM];  // 32 KB  K blocks row-major: Bt[key][d]
  __shared__ __bf16 Vt[DIM][BC];     // 16 KB  V block: Bt[d][key]
  __shared__ float  Sf[BR][BC];      // 8 KB
  __shared__ __bf16 Pb[BR][BC];      // 4 KB
  __shared__ float  rowm[BR], rowl[BR], rowscale[BR];

  const int tid  = threadIdx.x;
  const int wave = tid >> 5, lane = tid & 31;
  const int half = lane >> 4, nlane = lane & 15;
  const int b     = blockIdx.y;
  const int qbase = blockIdx.x * BR;
  const int wr = (wave & 3) * 16;    // row block (same for S and O)
  const int wc = (wave >> 2) * 128;  // O col block
  const int kc = (wave >> 2) * 16;   // S key-col block

  const unsigned ldsK0 = (unsigned)(size_t)(void*)&Ks[0][0][0];
  const unsigned ldsK1 = (unsigned)(size_t)(void*)&Ks[1][0][0];
  const unsigned ldsV  = (unsigned)(size_t)(void*)&Vt[0][0];

  // Preload this wave's Q fragments (16 rows x 256 dims) from global.
  const __bf16* qrow = Qb + (size_t)(b * NSEQ + qbase + wr) * DIM;
  v16bf aQ[8];
#pragma unroll
  for (int c = 0; c < 8; ++c) aQ[c] = load_frag_a(qrow + 32 * c, DIM, lane);

  v8f acc[8] = {};
  if (tid < BR) { rowm[tid] = -3.4e38f; rowl[tid] = 0.f; }

  const __bf16* kbase = Kb  + (size_t)b * NSEQ * DIM;
  const __bf16* vbase = Vtg + (size_t)b * DIM * NSEQ;
  const int NBLK = NSEQ / BC;        // 128

  // Prologue: DMA K block 0 (contiguous 32x256 bf16 = one 8192-elem row).
  if (wave == 0)
    tdm_load_2d_bf16(ldsK0, kbase, BC * DIM, 1, BC * DIM, 1, BC * DIM);

  for (int blk = 0; blk < NBLK; ++blk) {
    const int ib = blk & 1;
    if (wave == 0) __builtin_amdgcn_s_wait_tensorcnt(0);  // K_blk landed
    __syncthreads();                                      // Ks[ib] visible

    if (wave == 0) {
      // V block for this iteration: 256 rows x 32 keys, row stride NSEQ.
      tdm_load_2d_bf16(ldsV, vbase + (size_t)blk * BC,
                       BC, DIM, NSEQ, DIM, NSEQ);
      // Prefetch next K block into the other buffer.
      if (blk + 1 < NBLK)
        tdm_load_2d_bf16(ib ? ldsK0 : ldsK1,
                         kbase + (size_t)(blk + 1) * BC * DIM,
                         BC * DIM, 1, BC * DIM, 1, BC * DIM);
    }

    // S tile = Q(16x256) * K^T(256x16): 8 WMMAs over the 256-dim.
    v8f s = {};
#pragma unroll
    for (int c = 0; c < 8; ++c) {
      v16bf bk = load_frag_b(&Ks[ib][kc][32 * c], DIM, lane);
      s = __builtin_amdgcn_wmma_f32_16x16x32_bf16(
          false, aQ[c], false, bk, (short)0, s, false, false);
    }
#pragma unroll
    for (int v = 0; v < 8; ++v)
      Sf[wr + v + 8 * half][kc + nlane] = s[v];
    __syncthreads();

    // Online softmax: 4 lanes per query row (all 256 threads active).
    {
      int r  = tid >> 2;
      int j0 = (tid & 3) * 8;
      float mloc = -3.4e38f;
#pragma unroll
      for (int j = 0; j < 8; ++j) mloc = fmaxf(mloc, Sf[r][j0 + j]);
      mloc = fmaxf(mloc, __shfl_xor(mloc, 1));
      mloc = fmaxf(mloc, __shfl_xor(mloc, 2));
      float m_old = rowm[r];
      float mnew  = fmaxf(m_old, mloc);
      float sum = 0.f;
#pragma unroll
      for (int j = 0; j < 8; ++j) {
        float p = __expf(Sf[r][j0 + j] - mnew);
        sum += p;
        Pb[r][j0 + j] = f32_to_bf16(p);
      }
      sum += __shfl_xor(sum, 1);
      sum += __shfl_xor(sum, 2);
      if ((tid & 3) == 0) {
        float scale = __expf(m_old - mnew);
        rowl[r] = rowl[r] * scale + sum;
        rowm[r] = mnew;
        rowscale[r] = scale;
      }
    }
    // V_blk is the older of (<=2) outstanding tensor ops: wait to <=1.
    if (wave == 0) __builtin_amdgcn_s_wait_tensorcnt(1);
    __syncthreads();                                      // Pb + Vt visible

    // Rescale running output, then O += P(16x32) * V(32x128).
    float sc[8];
#pragma unroll
    for (int v = 0; v < 8; ++v) sc[v] = rowscale[wr + v + 8 * half];
#pragma unroll
    for (int t = 0; t < 8; ++t)
#pragma unroll
      for (int v = 0; v < 8; ++v) acc[t][v] *= sc[v];

    v16bf aP = load_frag_a(&Pb[wr][0], BC, lane);
#pragma unroll
    for (int t = 0; t < 8; ++t) {
      v16bf bv = load_frag_b(&Vt[wc + 16 * t][0], BC, lane);
      acc[t] = __builtin_amdgcn_wmma_f32_16x16x32_bf16(
          false, aP, false, bv, (short)0, acc[t], false, false);
    }
  }

  __syncthreads();
  if (tid < BR) rowscale[tid] = 1.0f / rowl[tid];
  __syncthreads();

  float invl[8];
#pragma unroll
  for (int v = 0; v < 8; ++v) invl[v] = rowscale[wr + v + 8 * half];
#pragma unroll
  for (int t = 0; t < 8; ++t) {
    int n = wc + 16 * t + nlane;
#pragma unroll
    for (int v = 0; v < 8; ++v) {
      size_t r = (size_t)(b * NSEQ + qbase + wr + v + 8 * half);
      ctx[r * DIM + n] = acc[t][v] * invl[v];
    }
  }
}

// ---------------------------------------------------------------------------
extern "C" void kernel_launch(void* const* d_in, const int* in_sizes, int n_in,
                              void* d_out, int out_size, void* d_ws, size_t ws_size,
                              hipStream_t stream) {
  (void)in_sizes; (void)n_in; (void)out_size; (void)ws_size;
  const float* query  = (const float*)d_in[0];
  const float* keyval = (const float*)d_in[1];
  const float* Wq = (const float*)d_in[2];
  const float* bq = (const float*)d_in[3];
  const float* Wk = (const float*)d_in[4];
  const float* bk = (const float*)d_in[5];
  const float* Wv = (const float*)d_in[6];
  const float* bv = (const float*)d_in[7];
  const float* Wo = (const float*)d_in[8];
  const float* bo = (const float*)d_in[9];

  char* ws = (char*)d_ws;
  const size_t bfMat = (size_t)MROWS * DIM * sizeof(__bf16);  // 8 MiB each
  __bf16* Qb  = (__bf16*)(ws);
  __bf16* Kb  = (__bf16*)(ws + bfMat);
  __bf16* Vtg = (__bf16*)(ws + 2 * bfMat);   // transposed per batch
  float*  ctx = (float*) (ws + 3 * bfMat);   // 16 MiB

  dim3 blk(256);
  gemm_bias_kernel<1><<<dim3(MROWS / GROWS), blk, 0, stream>>>(query,  Wq, bq, Qb,  MROWS);
  gemm_bias_kernel<1><<<dim3(MROWS / GROWS), blk, 0, stream>>>(keyval, Wk, bk, Kb,  MROWS);
  gemm_bias_kernel<2><<<dim3(MROWS / GROWS), blk, 0, stream>>>(keyval, Wv, bv, Vtg, MROWS);
  attn_kernel<<<dim3(NSEQ / BR, B_BATCH), blk, 0, stream>>>(Qb, Kb, Vtg, ctx);
  gemm_bias_kernel<0><<<dim3(MROWS / GROWS), blk, 0, stream>>>(ctx, Wo, bo, d_out, MROWS);
}
